// FP4Linear_19292993094262
// MI455X (gfx1250) — matrix operand order repaired and checked by reading, not compiled, not run on hardware
//
#include <hip/hip_runtime.h>

// ---------------------------------------------------------------------------
// FP4(e2m1)-weight linear layer:  y[M,N] = x[M,K] @ W[N,K]^T + bias[N]
//   x       : bf16 (raw 16-bit)                      [M*K]
//   w_packed: int32, one byte/slot, 2 FP4 codes/byte [N * K/2]
//   b_packed: int32, one byte/slot, 2 FP4 codes/byte [N/2]
//   out     : bf16                                   [M*N]
// Tiled WMMA GEMM (v_wmma_f32_16x16x32_bf16), f32 accum, double-buffered LDS,
// async global->LDS for activations, v_perm-LUT FP4 dequant.
// Block tile 256x128 (8 waves, 64x64 per wave -> 16 WMMA / K-step,
// 0.5KB LDS read per WMMA).
// ---------------------------------------------------------------------------

typedef __attribute__((ext_vector_type(16))) __bf16    v16bf;
typedef __attribute__((ext_vector_type(8)))  float     v8f;
typedef __attribute__((ext_vector_type(4)))  unsigned  v4u;

#define BM 256
#define BN 128
#define BK 32
#define LDS_STRIDE (BK + 8)   // halves per row, padded to dodge bank conflicts

union FragAB {
    v16bf v;
    v4u   u[2];
};

// ---- branch-free FP4(e2m1) -> packed bf16x2 via v_perm_b32 byte LUTs ------
// magnitudes (codes 0..7): 0, .5, 1, 1.5, 2, 3, 4, 6
//   bf16 bits:  0x0000 3F00 3F80 3FC0 4000 4040 4080 40C0
__device__ __forceinline__ unsigned fp4pair_to_bf16x2(unsigned b) {
    const unsigned sel = (b & 0x7u) | ((b & 0x70u) << 4);        // byte0=c3lo byte1=c3hi
    const unsigned hi2 = __builtin_amdgcn_perm(0x40404040u, 0x3F3F3F00u, sel);
    const unsigned lo2 = __builtin_amdgcn_perm(0xC0804000u, 0xC0800000u, sel);
    unsigned r = __builtin_amdgcn_perm(hi2, lo2, 0x05010400u);   // [hi1 lo1 hi0 lo0]
    r |= ((b & 0x8u) << 12) | ((b & 0x80u) << 24);               // sign bits 15 / 31
    return r;
}

__device__ __forceinline__ unsigned short f32_to_bf16_rne(float f) {
    unsigned u = __float_as_uint(f);
    unsigned r = 0x7FFFu + ((u >> 16) & 1u);
    return (unsigned short)((u + r) >> 16);
}

// generic LDS pointer -> 32-bit LDS offset (generic low 32 bits == DS address)
__device__ __forceinline__ unsigned lds_off(const void* p) {
    return (unsigned)(unsigned long long)(uintptr_t)p;
}

// async DMA-style copy: 16 bytes/lane global -> LDS, tracked with ASYNCcnt
__device__ __forceinline__ void async_copy_b128(unsigned lds_dst, const void* gsrc) {
    asm volatile("global_load_async_to_lds_b128 %0, %1, off"
                 :: "v"(lds_dst), "v"((unsigned long long)(uintptr_t)gsrc)
                 : "memory");
}
__device__ __forceinline__ void wait_asynccnt0() {
    asm volatile("s_wait_asynccnt 0x0" ::: "memory");
}

__global__ __launch_bounds__(256)
void fp4_linear_wmma_kernel(const unsigned short* __restrict__ x,   // bf16 bits [M][K]
                            const int*            __restrict__ wq,  // [N][K/2]
                            const int*            __restrict__ bq,  // [N/2]
                            unsigned short*       __restrict__ out, // bf16 bits [M][N]
                            int M, int N, int K)
{
    __shared__ unsigned short As[2][BM * LDS_STRIDE];   // 2 x 20KB
    __shared__ unsigned short Ws[2][BN * LDS_STRIDE];   // 2 x 10KB

    const int tid      = threadIdx.x;
    const int lane     = tid & 31;
    const int wid      = tid >> 5;          // 0..7
    const int laneRow  = lane & 15;
    const int laneHalf = lane >> 4;         // 0 or 1

    // 8 waves -> 4 (M) x 2 (N); each wave = 64x64 = 4x4 WMMA tiles
    const int wave_m = (wid >> 1) * 64;
    const int wave_n = (wid & 1) * 64;

    const int m0 = blockIdx.y * BM;
    const int n0 = blockIdx.x * BN;

    // staging mapping: 4 threads per row, passes of 64 rows
    const int grow = tid >> 2;              // 0..63
    const int acol = (tid & 3) * 8;         // halves within BK
    const int wcol = (tid & 3) * 4;         // int32 slots within BK/2

    const int K2 = K >> 1;                  // ints per weight row
    const int T  = K / BK;

    v4u wreg[2];
    v8f acc[4][4];
#pragma unroll
    for (int r = 0; r < 4; ++r)
#pragma unroll
        for (int c = 0; c < 4; ++c)
            acc[r][c] = (v8f){0.f, 0.f, 0.f, 0.f, 0.f, 0.f, 0.f, 0.f};

    // ---- prologue: tile 0 ----
#pragma unroll
    for (int p = 0; p < 4; ++p)
        async_copy_b128(lds_off(&As[0][(grow + p * 64) * LDS_STRIDE + acol]),
                        x + (size_t)(m0 + grow + p * 64) * K + acol);
#pragma unroll
    for (int p = 0; p < 2; ++p)
        wreg[p] = *(const v4u*)(wq + (size_t)(n0 + grow + p * 64) * K2 + wcol);
#pragma unroll
    for (int p = 0; p < 2; ++p) {
        v4u d;
        d.x = fp4pair_to_bf16x2(wreg[p].x);
        d.y = fp4pair_to_bf16x2(wreg[p].y);
        d.z = fp4pair_to_bf16x2(wreg[p].z);
        d.w = fp4pair_to_bf16x2(wreg[p].w);
        *(v4u*)(&Ws[0][(grow + p * 64) * LDS_STRIDE + wcol * 2]) = d;
    }
    wait_asynccnt0();
    __syncthreads();

    // ---- main K loop, double-buffered ----
    for (int t = 0; t < T; ++t) {
        const int cur = t & 1;

        // issue next tile's loads first (overlap with WMMA work below)
        if (t + 1 < T) {
            const int kk  = (t + 1) * BK;
            const int nxt = cur ^ 1;
#pragma unroll
            for (int p = 0; p < 4; ++p)
                async_copy_b128(lds_off(&As[nxt][(grow + p * 64) * LDS_STRIDE + acol]),
                                x + (size_t)(m0 + grow + p * 64) * K + kk + acol);
#pragma unroll
            for (int p = 0; p < 2; ++p)
                wreg[p] = *(const v4u*)(wq + (size_t)(n0 + grow + p * 64) * K2 +
                                        (kk >> 1) + wcol);
        }

        // A fragments: 16x32 bf16 per ISA layout
        //   lanes 0-15 : row = laneRow, K = {0..7, 16..23}
        //   lanes 16-31: row = laneRow, K = {8..15, 24..31}
        FragAB afrag[4];
#pragma unroll
        for (int r = 0; r < 4; ++r) {
            const unsigned short* base =
                &As[cur][(wave_m + r * 16 + laneRow) * LDS_STRIDE + laneHalf * 8];
            afrag[r].u[0] = *(const v4u*)(base);        // K 0..7  / 8..15
            afrag[r].u[1] = *(const v4u*)(base + 16);   // K 16..23 / 24..31
        }

        // B fragments: 32x16 bf16 (B[k][n] = W[n][k]); W-tile is n-major
        //   lanes 0-15 : col = laneRow, K = {0..7, 16..23} ; lanes 16-31: K = {8..15, 24..31}
        FragAB bfrag[4];
#pragma unroll
        for (int c = 0; c < 4; ++c) {
            const unsigned short* base =
                &Ws[cur][(wave_n + c * 16 + laneRow) * LDS_STRIDE + laneHalf * 8];
            bfrag[c].u[0] = *(const v4u*)(base);        // K 0..7  / 8..15
            bfrag[c].u[1] = *(const v4u*)(base + 16);   // K 16..23 / 24..31
        }

        // 16 WMMAs per K-step: 0.5KB of LDS reads per WMMA
#pragma unroll
        for (int r = 0; r < 4; ++r)
#pragma unroll
            for (int c = 0; c < 4; ++c)
                acc[r][c] = __builtin_amdgcn_wmma_f32_16x16x32_bf16(
                    false, afrag[r].v, false, bfrag[c].v,
                    (short)0, acc[r][c], false, false);

        // stage next W tile (dequantized) and publish both tiles
        if (t + 1 < T) {
            const int nxt = cur ^ 1;
#pragma unroll
            for (int p = 0; p < 2; ++p) {
                v4u d;
                d.x = fp4pair_to_bf16x2(wreg[p].x);
                d.y = fp4pair_to_bf16x2(wreg[p].y);
                d.z = fp4pair_to_bf16x2(wreg[p].z);
                d.w = fp4pair_to_bf16x2(wreg[p].w);
                *(v4u*)(&Ws[nxt][(grow + p * 64) * LDS_STRIDE + wcol * 2]) = d;
            }
            wait_asynccnt0();
            __syncthreads();
        }
    }

    // ---- epilogue: bias + bf16 store (non-temporal: write-once output) ----
    // C/D layout: VGPR i -> row = 8*laneHalf + i ; col = laneRow
#pragma unroll
    for (int c = 0; c < 4; ++c) {
        const int col = n0 + wave_n + c * 16 + laneRow;
        const unsigned pair = fp4pair_to_bf16x2((unsigned)bq[col >> 1] & 0xFFu);
        const unsigned bits = (col & 1) ? (pair >> 16) : (pair & 0xFFFFu);
        const float biasf   = __uint_as_float(bits << 16);
#pragma unroll
        for (int r = 0; r < 4; ++r) {
            const int rowBase = m0 + wave_m + r * 16 + laneHalf * 8;
#pragma unroll
            for (int i = 0; i < 8; ++i) {
                const float v = acc[r][c][i] + biasf;
                __builtin_nontemporal_store(
                    f32_to_bf16_rne(v),
                    out + (size_t)(rowBase + i) * N + col);
            }
        }
    }
}

extern "C" void kernel_launch(void* const* d_in, const int* in_sizes, int n_in,
                              void* d_out, int out_size, void* d_ws, size_t ws_size,
                              hipStream_t stream) {
    const unsigned short* x   = (const unsigned short*)d_in[0]; // bf16 bits
    const int*            wq  = (const int*)d_in[1];
    const int*            bq  = (const int*)d_in[2];
    unsigned short*       out = (unsigned short*)d_out;

    const int N = in_sizes[2] * 2;                               // OUT
    const int K = (int)(((long long)in_sizes[1] * 2) / N);       // IN
    const int M = (int)((long long)in_sizes[0] / K);             // B*S

    dim3 grid(N / BN, M / BM);
    fp4_linear_wmma_kernel<<<grid, 256, 0, stream>>>(x, wq, bq, out, M, N, K);
}
